// SoftMinLayer_12953621364967
// MI455X (gfx1250) — matrix operand order repaired and checked by reading, not compiled, not run on hardware
//
#include <hip/hip_runtime.h>
#include <hip/hip_bf16.h>

typedef __attribute__((ext_vector_type(16))) _Float16 v16h;
typedef __attribute__((ext_vector_type(8)))  float    v8f;

#define NSEQ    512
#define QLEN    1024
#define KSH     100
#define LSH     50
#define JW      (QLEN - LSH)        // 974 windows
#define JT      ((JW + 15) / 16)    // 61 M-tiles
#define NT      ((KSH + 15) / 16)   // 7 N-tiles
#define NTHREADS (NT * 32)          // 224 threads = 7 waves (wave32)
#define SEQ_PAD 1088                // covers base+KA max = 960+15+63 = 1038
#define CEXP    (-2.0f)             // ALPHA / L = -100/50

__device__ __forceinline__
v8f tile_cross(const _Float16* __restrict__ sseq, int jt, int m, int h,
               v16h b0, v16h b1)
{
    const int base = jt * 16 + m;   // window index of this lane's A row
    // A layout (16x32, f16): M = lane&15,
    // slot (v,half) -> K = 16*(v/4) + 8*h + 2*(v%4) + half
    // => two contiguous 8-half runs per v16h (compiler emits ds_load_b128).
    v16h a0, a1;
    #pragma unroll
    for (int v = 0; v < 8; ++v) {
        #pragma unroll
        for (int q = 0; q < 2; ++q) {
            int KA = 16 * (v >> 2) + 8 * h + 2 * (v & 3) + q;
            a0[2 * v + q] = sseq[base + KA];
            a1[2 * v + q] = sseq[base + 32 + KA];
        }
    }
    v8f c = {};
    c = __builtin_amdgcn_wmma_f32_16x16x32_f16(false, a0, false, b0,
                                               (short)0, c, false, false);
    c = __builtin_amdgcn_wmma_f32_16x16x32_f16(false, a1, false, b1,
                                               (short)0, c, false, false);
    return c;
}

__device__ __forceinline__
void tile_min(v8f c, const float* __restrict__ ssq, int jb,
              float& rmin0, float& rmin1)
{
    #pragma unroll
    for (int r = 0; r < 8; r += 2) {
        float t0 = fmaf(-2.0f, c[r],     ssq[jb + r]);
        float t1 = fmaf(-2.0f, c[r + 1], ssq[jb + r + 1]);
        rmin0 = fminf(rmin0, t0);
        rmin1 = fminf(rmin1, t1);
    }
}

__device__ __forceinline__
void tile_acc(v8f c, const float* __restrict__ ssq, int jb, float rmin,
              float& S1a, float& S1b, float& S2a, float& S2b)
{
    #pragma unroll
    for (int r = 0; r < 8; r += 2) {
        float u0 = fmaf(-2.0f, c[r],     ssq[jb + r])     - rmin;
        float u1 = fmaf(-2.0f, c[r + 1], ssq[jb + r + 1]) - rmin;
        float e0 = __expf(CEXP * u0);
        float e1 = __expf(CEXP * u1);
        S1a += e0;
        S1b += e1;
        S2a = fmaf(u0, e0, S2a);
        S2b = fmaf(u1, e1, S2b);
    }
}

__global__ __launch_bounds__(NTHREADS)
void softmin_shapelet_kernel(const float* __restrict__ in,
                             const float* __restrict__ shp,
                             float* __restrict__ out)
{
    __shared__ _Float16 sseq[SEQ_PAD];   // f16 sequence, zero padded (WMMA A src)
    __shared__ float    sseqf[QLEN];     // f32 sequence (window energies)
    __shared__ float    ssq[JT * 16];    // per-window sum of squares (pad=1e30)

    const int tid  = threadIdx.x;
    const int wave = tid >> 5;
    const int lane = tid & 31;
    const int h    = lane >> 4;          // half-wave select
    const int m    = lane & 15;
    const int seq  = blockIdx.x;

    // Stage the sequence row (4 KB, L2-resident) into LDS: f32 + f16 copies.
    for (int t = tid; t < QLEN; t += NTHREADS) {
        float x = in[seq * QLEN + t];
        sseqf[t] = x;
        sseq[t]  = (_Float16)x;
    }
    for (int t = QLEN + tid; t < SEQ_PAD; t += NTHREADS)
        sseq[t] = (_Float16)0.0f;
    __syncthreads();

    // Sliding-window energies; poisoned past JW so padded rows weight to 0.
    for (int j = tid; j < JT * 16; j += NTHREADS) {
        float s;
        if (j < JW) {
            s = 0.0f;
            for (int t = 0; t < LSH; ++t) {
                float x = sseqf[j + t];
                s = fmaf(x, x, s);
            }
        } else {
            s = 1.0e30f;
        }
        ssq[j] = s;
    }
    __syncthreads();

    // Each wave owns one 16-shapelet column tile; B stays resident in VGPRs.
    // B layout (32x16, f16): same K order as A,
    // slot (v,half) -> K = 16*(v/4) + 8*h + 2*(v%4) + half; N = lane&15.
    const int kcol = wave * 16 + m;
    v16h b0, b1;
    #pragma unroll
    for (int v = 0; v < 8; ++v) {
        #pragma unroll
        for (int q = 0; q < 2; ++q) {
            int KB = 16 * (v >> 2) + 8 * h + 2 * (v & 3) + q; // 0..31 (< LSH always)
            float x0 = (kcol < KSH) ? shp[kcol * LSH + KB] : 0.0f;
            float x1 = (kcol < KSH && (KB + 32) < LSH) ? shp[kcol * LSH + KB + 32] : 0.0f;
            b0[2 * v + q] = (_Float16)x0;
            b1[2 * v + q] = (_Float16)x1;
        }
    }

    // ---- Pass 1: column-wise min of t_j = ssq[j] - 2*cross_j ----------------
    // (s_sq/L shift and 1/L scale cancel out of the softmin weights.)
    // 2-stage software pipeline: issue tile jt's WMMAs, then consume tile
    // jt-1's accumulator — the independent epilogue VALU fills the WMMA->VALU
    // hazard slots (no v_nops) and hides the LDS A-gather latency.
    float rmin0 = 3.0e38f, rmin1 = 3.0e38f;
    {
        v8f c = tile_cross(sseq, 0, m, h, b0, b1);
        for (int jt = 1; jt < JT; ++jt) {       // 60 trips (even)
            v8f cn = tile_cross(sseq, jt, m, h, b0, b1);
            tile_min(c, ssq, (jt - 1) * 16 + 8 * h, rmin0, rmin1);
            c = cn;
        }
        tile_min(c, ssq, (JT - 1) * 16 + 8 * h, rmin0, rmin1);
    }
    float rmin = fminf(rmin0, rmin1);
    rmin = fminf(rmin, __shfl_xor(rmin, 16, 32));   // global column min

    // ---- Pass 2: softmin-weighted accumulation ------------------------------
    // u = t_j - rmin >= 0 ; w = exp(ALPHA/L * u) ; S1 = sum w ; S2 = sum u*w.
    float S1a = 0.0f, S1b = 0.0f, S2a = 0.0f, S2b = 0.0f;
    {
        v8f c = tile_cross(sseq, 0, m, h, b0, b1);
        for (int jt = 1; jt < JT; ++jt) {       // 60 trips (even)
            v8f cn = tile_cross(sseq, jt, m, h, b0, b1);
            tile_acc(c, ssq, (jt - 1) * 16 + 8 * h, rmin, S1a, S1b, S2a, S2b);
            c = cn;
        }
        tile_acc(c, ssq, (JT - 1) * 16 + 8 * h, rmin, S1a, S1b, S2a, S2b);
    }
    float S1 = S1a + S1b;
    float S2 = S2a + S2b;

    // Column kcol is split across the lane pair (lane, lane^16): same rmin on
    // both sides, so the merge is a plain add.
    S1 += __shfl_xor(S1, 16, 32);
    S2 += __shfl_xor(S2, 16, 32);

    // M = sum(Dm*w)/sum(w) with Dm = u/L  =>  M = S2 / (S1 * L).
    if (h == 0 && kcol < KSH)
        out[seq * KSH + kcol] = S2 / (S1 * (float)LSH);
}

extern "C" void kernel_launch(void* const* d_in, const int* in_sizes, int n_in,
                              void* d_out, int out_size, void* d_ws, size_t ws_size,
                              hipStream_t stream) {
    const float* in  = (const float*)d_in[0];   // (512, 1024) f32
    const float* shp = (const float*)d_in[1];   // (100, 50)  f32
    float* out = (float*)d_out;                 // (512, 100) f32
    (void)in_sizes; (void)n_in; (void)out_size; (void)d_ws; (void)ws_size;

    softmin_shapelet_kernel<<<NSEQ, NTHREADS, 0, stream>>>(in, shp, out);
}
